// AspectAttention_83966610636803
// MI455X (gfx1250) — matrix operand order repaired and checked by reading, not compiled, not run on hardware
//
#include <hip/hip_runtime.h>
#include <hip/hip_bf16.h>

// Problem sizes (fixed by the reference)
#define BB 16
#define SS 2048
#define HH 1024
#define AA 8

#define MB 128            // token rows per block in the score GEMM
#define KPAD 8            // bf16 elements of LDS row padding (16B) to break bank conflicts
#define AST (HH + KPAD)   // LDS row stride in bf16 elements

typedef __attribute__((ext_vector_type(16))) __bf16 v16bf;
typedef __attribute__((ext_vector_type(8)))  __bf16 v8bf;
typedef __attribute__((ext_vector_type(8)))  float  v8f;

__device__ __forceinline__ __bf16 f2bf(float f) {
  union { float f; unsigned u; } x; x.f = f;
  unsigned r = x.u + 0x7FFFu + ((x.u >> 16) & 1u);   // round-to-nearest-even
  unsigned short h = (unsigned short)(r >> 16);
  return __builtin_bit_cast(__bf16, h);
}

// Hardware tanh (V_TANH_F32 on gfx1250) with safe fallbacks.
__device__ __forceinline__ float htanh(float x) {
#if __has_builtin(__builtin_amdgcn_tanhf)
  return __builtin_amdgcn_tanhf(x);
#elif __has_builtin(__builtin_amdgcn_tanh_f32)
  return __builtin_amdgcn_tanh_f32(x);
#else
  return tanhf(x);
#endif
}

// ---------------------------------------------------------------------------
// Prep 1: mean over aspect tokens  aspm[b][h] = (1/A) sum_a aspect[b,a,h]
__global__ void aspmean_k(const float* __restrict__ asp, float* __restrict__ aspm) {
  int i = blockIdx.x * 256 + threadIdx.x;            // 16384 = B*H
  int b = i / HH, h = i % HH;
  float s = 0.f;
#pragma unroll
  for (int a = 0; a < AA; ++a)
    s += asp[((size_t)b * AA + a) * HH + h];
  aspm[i] = s * (1.0f / AA);
}

// Prep 2: aspP[b][h] = aspm[b,:] @ W2[:,h] + bias[h]    (W2 = W[H:2H, :])
__global__ void aspproj_k(const float* __restrict__ aspm, const float* __restrict__ W,
                          const float* __restrict__ bias, float* __restrict__ aspP) {
  int i = blockIdx.x * 256 + threadIdx.x;            // 16384 = B*H
  int b = i / HH, h = i % HH;
  const float* am = aspm + (size_t)b * HH;
  float s = bias[h];
  for (int k = 0; k < HH; ++k)
    s += am[k] * W[((size_t)(HH + k)) * HH + h];     // coalesced over h
  aspP[i] = s;
}

// Prep 3: Wt[n][k] = bf16(W1[k][n])  -- transposed W1 so B-fragments are contiguous-K
__global__ void wtrans_k(const float* __restrict__ W, __bf16* __restrict__ Wt) {
  __shared__ float tile[32][33];
  int bx = blockIdx.x & 31;        // n tile
  int by = blockIdx.x >> 5;        // k tile
  int tx = threadIdx.x & 31;
  int ty = threadIdx.x >> 5;       // 0..7
  for (int yy = ty; yy < 32; yy += 8)
    tile[yy][tx] = W[((size_t)(by * 32 + yy)) * HH + bx * 32 + tx];
  __syncthreads();
  for (int yy = ty; yy < 32; yy += 8)
    Wt[((size_t)(bx * 32 + yy)) * HH + by * 32 + tx] = f2bf(tile[tx][yy]);
}

// ---------------------------------------------------------------------------
// Main: scores[row] = sum_h v[h] * tanh( token[row,:] @ W1[:,h] + aspP[batch,h] )
// Block: 256 threads = 8 waves; each wave owns one 16-row M tile.
__global__ __launch_bounds__(256)
void score_gemm_k(const float* __restrict__ tok, const __bf16* __restrict__ Wt,
                  const float* __restrict__ aspP, const float* __restrict__ v,
                  float* __restrict__ scores) {
  __shared__ __bf16 Alds[MB * AST];                  // 128 x 1032 bf16 = 258 KB

  const int row0  = blockIdx.x * MB;                 // flattened b*s row
  const int batch = row0 / SS;
  const int tid   = threadIdx.x;

  // Stage A: 128 rows x 1024 cols fp32 -> bf16 LDS (float4 loads)
  for (int i = tid; i < MB * HH / 4; i += 256) {
    int r  = i >> 8;                                 // / (H/4)
    int c4 = i & 255;
    float4 f = ((const float4*)(tok + (size_t)(row0 + r) * HH))[c4];
    __bf16* dst = &Alds[r * AST + c4 * 4];
    dst[0] = f2bf(f.x); dst[1] = f2bf(f.y); dst[2] = f2bf(f.z); dst[3] = f2bf(f.w);
  }
  __syncthreads();

  const int wave = tid >> 5;
  const int lane = tid & 31;
  const int half = lane >> 4;                        // 0: lanes 0-15, 1: lanes 16-31
  const int r16  = lane & 15;
  const __bf16* arow = &Alds[(wave * 16 + r16) * AST];

  float p[8];
#pragma unroll
  for (int j = 0; j < 8; ++j) p[j] = 0.f;

  for (int n0 = 0; n0 < HH; n0 += 16) {
    const int col = n0 + r16;
    const __bf16* brow = Wt + (size_t)col * HH;
    v8f acc0 = {};
    v8f acc1 = {};
    // Two independent accumulation chains (K chunks kk and kk+32) so the
    // matrix pipe always has a WMMA whose C input is ready.
#pragma unroll 2
    for (int kk = 0; kk < HH; kk += 64) {
      // A fragment (16x32 bf16): lanes 0-15 rows, VGPR0-3 = K +0..7 (lo) / +8..15 (hi),
      //                          VGPR4-7 = K +16..23 (lo) / +24..31 (hi)
      v8bf alo0 = *(const v8bf*)(arow + kk + half * 8);
      v8bf ahi0 = *(const v8bf*)(arow + kk + 16 + half * 8);
      v16bf a0 = __builtin_shufflevector(alo0, ahi0, 0,1,2,3,4,5,6,7,8,9,10,11,12,13,14,15);
      v8bf alo1 = *(const v8bf*)(arow + kk + 32 + half * 8);
      v8bf ahi1 = *(const v8bf*)(arow + kk + 48 + half * 8);
      v16bf a1 = __builtin_shufflevector(alo1, ahi1, 0,1,2,3,4,5,6,7,8,9,10,11,12,13,14,15);
      // B fragments (32x16 bf16): lane = column, same K order as A
      v8bf blo0 = *(const v8bf*)(brow + kk + half * 8);
      v8bf bhi0 = *(const v8bf*)(brow + kk + 16 + half * 8);
      v16bf b0 = __builtin_shufflevector(blo0, bhi0, 0,1,2,3,4,5,6,7,8,9,10,11,12,13,14,15);
      v8bf blo1 = *(const v8bf*)(brow + kk + 32 + half * 8);
      v8bf bhi1 = *(const v8bf*)(brow + kk + 48 + half * 8);
      v16bf b1 = __builtin_shufflevector(blo1, bhi1, 0,1,2,3,4,5,6,7,8,9,10,11,12,13,14,15);
      acc0 = __builtin_amdgcn_wmma_f32_16x16x32_bf16(
          false, a0, false, b0, (short)0, acc0, false, false);
      acc1 = __builtin_amdgcn_wmma_f32_16x16x32_bf16(
          false, a1, false, b1, (short)0, acc1, false, false);
    }
    // C tile: VGPR j holds (M = j + 8*half, N = r16)
    const float av = aspP[batch * HH + col];
    const float vv = v[col];
#pragma unroll
    for (int j = 0; j < 8; ++j)
      p[j] += vv * htanh(acc0[j] + acc1[j] + av);
  }

  // Reduce over the 16 columns: xor 1,2,4,8 stays within each 16-lane half.
#pragma unroll
  for (int m = 1; m <= 8; m <<= 1) {
#pragma unroll
    for (int j = 0; j < 8; ++j)
      p[j] += __shfl_xor(p[j], m, 32);
  }
  if (r16 == 0) {
    int rbase = row0 + wave * 16 + half * 8;
#pragma unroll
    for (int j = 0; j < 8; ++j)
      scores[rbase + j] = p[j];
  }
}

// ---------------------------------------------------------------------------
// Softmax over S per batch: weights[b,s] = softmax(scores[b,:])[s]
__global__ void softmax_k(const float* __restrict__ scores, float* __restrict__ weights) {
  __shared__ float sh[256];
  const int b = blockIdx.x;
  const int tid = threadIdx.x;
  const float* s = scores + (size_t)b * SS;

  float m = -3.4e38f;
  for (int i = tid; i < SS; i += 256) m = fmaxf(m, s[i]);
  sh[tid] = m; __syncthreads();
  for (int off = 128; off > 0; off >>= 1) {
    if (tid < off) sh[tid] = fmaxf(sh[tid], sh[tid + off]);
    __syncthreads();
  }
  m = sh[0]; __syncthreads();

  float sum = 0.f;
  for (int i = tid; i < SS; i += 256) sum += __expf(s[i] - m);
  sh[tid] = sum; __syncthreads();
  for (int off = 128; off > 0; off >>= 1) {
    if (tid < off) sh[tid] += sh[tid + off];
    __syncthreads();
  }
  const float inv = 1.0f / sh[0];
  for (int i = tid; i < SS; i += 256)
    weights[(size_t)b * SS + i] = __expf(s[i] - m) * inv;
}

// Final: out[row, h] = tok[row, h] * (1 + weights[row])   (float4 per thread)
__global__ void scale_k(const float* __restrict__ tok, const float* __restrict__ w,
                        float* __restrict__ out) {
  size_t i = (size_t)blockIdx.x * 256 + threadIdx.x;   // one float4 each
  const float4 t = ((const float4*)tok)[i];
  int row = (int)(i >> 8);                             // (i*4)/H, H=1024
  float sc = 1.0f + w[row];
  float4 o; o.x = t.x * sc; o.y = t.y * sc; o.z = t.z * sc; o.w = t.w * sc;
  ((float4*)out)[i] = o;
}

// ---------------------------------------------------------------------------
extern "C" void kernel_launch(void* const* d_in, const int* in_sizes, int n_in,
                              void* d_out, int out_size, void* d_ws, size_t ws_size,
                              hipStream_t stream) {
  const float* tok  = (const float*)d_in[0];   // [B,S,H]
  const float* asp  = (const float*)d_in[1];   // [B,A,H]
  const float* W    = (const float*)d_in[2];   // [2H,H]
  const float* bias = (const float*)d_in[3];   // [H]
  const float* v    = (const float*)d_in[4];   // [H]
  float* out = (float*)d_out;

  char* ws = (char*)d_ws;
  __bf16* Wt     = (__bf16*)ws;                              // 2 MB
  float*  aspm   = (float*)(ws + (size_t)2 * 1024 * 1024);   // 64 KB
  float*  aspP   = aspm + BB * HH;                           // 64 KB
  float*  scores = aspP + BB * HH;                           // 128 KB
  float*  wgt    = scores + BB * SS;                         // 128 KB

  // Prep (tiny)
  wtrans_k <<<1024, 256, 0, stream>>>(W, Wt);
  aspmean_k<<<(BB * HH) / 256, 256, 0, stream>>>(asp, aspm);
  aspproj_k<<<(BB * HH) / 256, 256, 0, stream>>>(aspm, W, bias, aspP);

  // Main bf16 WMMA score GEMM: 32768 rows / 128 per block
  score_gemm_k<<<(BB * SS) / MB, 256, 0, stream>>>(tok, Wt, aspP, v, scores);

  // Softmax + final scale
  softmax_k<<<BB, 256, 0, stream>>>(scores, wgt);
  scale_k  <<<(size_t)(BB * SS) * HH / 4 / 256, 256, 0, stream>>>(tok, wgt, out);
}
